// E63dResidualNonlinearCell_4887672783121
// MI455X (gfx1250) — matrix operand order repaired and checked by reading, not compiled, not run on hardware
//
#include <hip/hip_runtime.h>

// ---------------------------------------------------------------------------
// Residual nonlinear RNN cell for MI455X (gfx1250, wave32, WMMA).
//   Phase 1: batched input GEMMs (alpha_x, wx) with v_wmma_f32_16x16x32_bf16.
//   Phase 2: persistent-kernel sequential scan; W_h slice LDS-resident per WGP
//            (64 KB static LDS), h state in registers, device-wide atomic
//            barrier per timestep, next-step operand prefetch.
// ---------------------------------------------------------------------------

typedef __attribute__((ext_vector_type(16))) __bf16 v16bf;
typedef __attribute__((ext_vector_type(8)))  __bf16 v8bf;
typedef __attribute__((ext_vector_type(8)))  float  v8f;
typedef __attribute__((ext_vector_type(4)))  float  v4f;

#define T_STEPS 512
#define BATCH   32
#define DIM     1024
#define TBROWS  (T_STEPS * BATCH)   // 16384 rows of X
#define BD      (BATCH * DIM)       // 32768
#define NWG_SCAN 32                 // persistent workgroups in the scan

static __device__ __forceinline__ float fast_sigmoid(float x) {
  return 1.0f / (1.0f + __expf(-x));
}
static __device__ __forceinline__ float fast_tanh(float x) {
  return 1.0f - 2.0f / (1.0f + __expf(2.0f * x));
}

// A-fragment (16x32 bf16, MxK): lane m holds K = koff..koff+15 (koff = 16*half).
// Two contiguous 16B chunks -> global_load_b128 pair.
static __device__ __forceinline__ v16bf load_frag_a(const __bf16* p) {
  v8bf lo = *(const v8bf*)(p);
  v8bf hi = *(const v8bf*)(p + 8);
  return __builtin_shufflevector(lo, hi, 0, 1, 2, 3, 4, 5, 6, 7,
                                         8, 9, 10, 11, 12, 13, 14, 15);
}

static __device__ __forceinline__ v8f wmma_bf16(v16bf a, v16bf b, v8f c) {
  return __builtin_amdgcn_wmma_f32_16x16x32_bf16(
      /*neg_a=*/false, a, /*neg_b=*/false, b,
      /*c_mod=*/(short)0, c, /*reuse_a=*/false, /*reuse_b=*/false);
}

// ---------------------------------------------------------------------------
// f32 -> bf16 conversion, 4 elements / thread
// ---------------------------------------------------------------------------
__global__ __launch_bounds__(256) void cvt4_kernel(const float* __restrict__ src,
                                                   __bf16* __restrict__ dst, int n4) {
  int i = blockIdx.x * blockDim.x + threadIdx.x;
  if (i < n4) {
    v4f f = *(const v4f*)(src + (size_t)i * 4);
    __bf16* d = dst + (size_t)i * 4;
    d[0] = (__bf16)f.x; d[1] = (__bf16)f.y; d[2] = (__bf16)f.z; d[3] = (__bf16)f.w;
  }
}

// ---------------------------------------------------------------------------
// init: h[0] = h0 (f32, into d_out h section), hbuf[0] = bf16(h0), counter = 0
// ---------------------------------------------------------------------------
__global__ __launch_bounds__(256) void init_kernel(const float* __restrict__ h0,
                                                   float* __restrict__ h_sec,
                                                   __bf16* __restrict__ hbuf0,
                                                   unsigned* __restrict__ counter) {
  int i = blockIdx.x * blockDim.x + threadIdx.x;
  if (i < BD) {
    float v = h0[i];
    h_sec[i] = v;
    hbuf0[i] = (__bf16)v;
  }
  if (i == 0) *counter = 0u;
}

// ---------------------------------------------------------------------------
// Phase 1: alpha_x = X Wa^T + b_alpha ; wx = X Wx^T
// One wave computes a 16(M) x 64(N) strip with K=1024.
// wave id bits: [0:3]=ngroup(16), [4:13]=mtile(1024), [14]=matrix select.
// ---------------------------------------------------------------------------
__global__ __launch_bounds__(256) void input_gemm_kernel(
    const __bf16* __restrict__ xb,
    const __bf16* __restrict__ wa_b,
    const __bf16* __restrict__ wx_b,
    const float*  __restrict__ b_alpha,
    float* __restrict__ alpha_out,
    float* __restrict__ wx_out) {
  const int wave = (int)((blockIdx.x * 256u + threadIdx.x) >> 5);
  const int lane = threadIdx.x & 31;
  const int lm   = lane & 15;
  const int half = lane >> 4;

  const int ngroup = wave & 15;
  const int mtile  = (wave >> 4) & 1023;
  const int mat    = wave >> 14;   // 0 = alpha, 1 = wx

  const __bf16* __restrict__ w = mat ? wx_b : wa_b;
  const int nbase = ngroup * 64;

  const __bf16* abase = xb + (size_t)(mtile * 16 + lm) * DIM + half * 16;
  const __bf16* wb0 = w + (size_t)(nbase +  0 + lm) * DIM + half * 16;
  const __bf16* wb1 = w + (size_t)(nbase + 16 + lm) * DIM + half * 16;
  const __bf16* wb2 = w + (size_t)(nbase + 32 + lm) * DIM + half * 16;
  const __bf16* wb3 = w + (size_t)(nbase + 48 + lm) * DIM + half * 16;

  v8f acc0 = {}, acc1 = {}, acc2 = {}, acc3 = {};
#pragma unroll 4
  for (int kc = 0; kc < 32; ++kc) {
    v16bf a = load_frag_a(abase + kc * 32);
    acc0 = wmma_bf16(a, *(const v16bf*)(wb0 + kc * 32), acc0);
    acc1 = wmma_bf16(a, *(const v16bf*)(wb1 + kc * 32), acc1);
    acc2 = wmma_bf16(a, *(const v16bf*)(wb2 + kc * 32), acc2);
    acc3 = wmma_bf16(a, *(const v16bf*)(wb3 + kc * 32), acc3);
  }

  float* __restrict__ outp = mat ? wx_out : alpha_out;
  float bv0 = 0.f, bv1 = 0.f, bv2 = 0.f, bv3 = 0.f;
  if (!mat) {
    bv0 = b_alpha[nbase +  0 + lm];
    bv1 = b_alpha[nbase + 16 + lm];
    bv2 = b_alpha[nbase + 32 + lm];
    bv3 = b_alpha[nbase + 48 + lm];
  }
#pragma unroll
  for (int r = 0; r < 8; ++r) {
    const size_t row = (size_t)(mtile * 16 + r + half * 8) * DIM;
    outp[row + nbase +  0 + lm] = acc0[r] + bv0;
    outp[row + nbase + 16 + lm] = acc1[r] + bv1;
    outp[row + nbase + 32 + lm] = acc2[r] + bv2;
    outp[row + nbase + 48 + lm] = acc3[r] + bv3;
  }
}

// ---------------------------------------------------------------------------
// Phase 2: persistent scan. 32 WGs x 128 threads (4 wave32). WG wg owns
// columns [wg*32, wg*32+32). Its W_h slice (32x1024 bf16 = 64 KB) lives in
// static LDS in WMMA B-fragment order (conflict-free 32B/lane reads).
// Each wave owns one 16x16 output tile for the whole scan; its f32 h-state
// stays in registers. alpha_x/wx loads are hoisted ahead of the WMMA chain
// and the next step's rows are prefetched (global_prefetch_b8).
// ---------------------------------------------------------------------------
__global__ __launch_bounds__(128) void scan_kernel(
    const float* __restrict__ Wh,
    const float* __restrict__ bias,
    const float* __restrict__ alpha_x,
    const float* __restrict__ wx,
    const float* __restrict__ h0,
    float* __restrict__ out_sec,     // [T,B,D]
    float* __restrict__ h_sec,       // [T+1,B,D]
    __bf16* __restrict__ hbuf,       // [2][B*D] bf16 double buffer
    unsigned* __restrict__ counter) {
  __shared__ __bf16 wh_lds[32 * 2 * 32 * 16];   // [kc 32][ntile 2][lane 32][16] = 64 KB

  const int wg   = blockIdx.x;       // 0..31
  const int wv   = threadIdx.x >> 5; // 0..3
  const int lane = threadIdx.x & 31;
  const int lm   = lane & 15;
  const int half = lane >> 4;

  const int nbase = wg * 32;
  const int mtile = wv >> 1;         // 0..1  (M = 32)
  const int ntile = wv & 1;          // 0..1  (N slice of 32)

  // ---- preload W_h slice into LDS, pre-swizzled into B-fragment order ----
  for (int i = 0; i < 8; ++i) {
    const int kc = wv * 8 + i;
#pragma unroll
    for (int nt = 0; nt < 2; ++nt) {
      const int e = nbase + nt * 16 + lm;                 // output column
      const v4f* s4 = (const v4f*)(Wh + (size_t)e * DIM + kc * 32 + half * 16);
      v16bf v;
#pragma unroll
      for (int q = 0; q < 4; ++q) {
        v4f f = s4[q];
        v[q * 4 + 0] = (__bf16)f.x; v[q * 4 + 1] = (__bf16)f.y;
        v[q * 4 + 2] = (__bf16)f.z; v[q * 4 + 3] = (__bf16)f.w;
      }
      *(v16bf*)(wh_lds + ((size_t)(kc * 2 + nt) * 32 + lane) * 16) = v;
    }
  }
  __syncthreads();

  // ---- per-wave persistent f32 h state for this tile ----
  const int n = nbase + ntile * 16 + lm;    // this lane's column
  float hreg[8];
#pragma unroll
  for (int r = 0; r < 8; ++r) {
    const int m = mtile * 16 + r + half * 8;
    hreg[r] = h0[(size_t)m * DIM + n];
  }
  const float bn = bias[n];
  const int arow = mtile * 16 + lm;
  const __bf16* lb = wh_lds + ((size_t)ntile * 32 + lane) * 16;  // +1024 elems / kc

  for (int t = 0; t < T_STEPS; ++t) {
    const __bf16* hcur  = hbuf + (size_t)(t & 1) * BD;
    __bf16*       hnext = hbuf + (size_t)((t + 1) & 1) * BD;
    const __bf16* ab = hcur + (size_t)arow * DIM + half * 16;

    // Hoisted per-step operands (independent of h): issue before WMMA chain.
    const size_t tb = (size_t)t * BD;
    float axv[8], wxv[8];
#pragma unroll
    for (int r = 0; r < 8; ++r) {
      const int m = mtile * 16 + r + half * 8;
      const size_t idx = tb + (size_t)m * DIM + n;
      axv[r] = alpha_x[idx];
      wxv[r] = wx[idx];
    }

    v8f acc = {};
#pragma unroll 4
    for (int kc = 0; kc < 32; ++kc) {
      v16bf a = load_frag_a(ab + kc * 32);
      v16bf b = *(const v16bf*)(lb + (size_t)kc * 1024);
      acc = wmma_bf16(a, b, acc);
    }

    // Prefetch next step's alpha_x/wx rows into cache while this step drains.
    const size_t tbn = (size_t)((t + 1 < T_STEPS) ? t + 1 : t) * BD;
#pragma unroll
    for (int r = 0; r < 8; ++r) {
      const int m = mtile * 16 + r + half * 8;
      const size_t idxn = tbn + (size_t)m * DIM + n;
      __builtin_prefetch(&alpha_x[idxn], 0, 0);
      __builtin_prefetch(&wx[idxn], 0, 0);
    }

#pragma unroll
    for (int r = 0; r < 8; ++r) {
      const int m = mtile * 16 + r + half * 8;
      const size_t idx = tb + (size_t)m * DIM + n;
      const float al = fast_sigmoid(axv[r]);
      const float vv = fast_tanh(acc[r] + wxv[r] + bn);
      const float hn = hreg[r] + al * vv;
      hreg[r] = hn;
      h_sec[idx + BD] = hn;                       // h[t+1]
      out_sec[idx] = hn * hn * fast_sigmoid(hn);  // h * silu(h)
      hnext[(size_t)m * DIM + n] = (__bf16)hn;
    }

    // ---- device-wide barrier (monotonic target, reset by init each call) ----
    __syncthreads();
    if (threadIdx.x == 0) {
      __hip_atomic_fetch_add(counter, 1u, __ATOMIC_RELEASE, __HIP_MEMORY_SCOPE_AGENT);
      const unsigned target = (unsigned)(t + 1) * NWG_SCAN;
      while (__hip_atomic_load(counter, __ATOMIC_ACQUIRE, __HIP_MEMORY_SCOPE_AGENT) < target)
        __builtin_amdgcn_s_sleep(1);
    }
    __syncthreads();
  }
}

// ---------------------------------------------------------------------------
// Host launcher
// Inputs: x[T,B,D], h0[B,D], W_alpha[D,D], b_alpha[D], W_h[D,D], W_x[D,D], b[D]
// d_out = output[T,B,D] ++ h[T+1,B,D]  (f32)
// ---------------------------------------------------------------------------
extern "C" void kernel_launch(void* const* d_in, const int* in_sizes, int n_in,
                              void* d_out, int out_size, void* d_ws, size_t ws_size,
                              hipStream_t stream) {
  (void)in_sizes; (void)n_in; (void)out_size; (void)ws_size;

  const float* x       = (const float*)d_in[0];
  const float* h0      = (const float*)d_in[1];
  const float* W_alpha = (const float*)d_in[2];
  const float* b_alpha = (const float*)d_in[3];
  const float* W_h     = (const float*)d_in[4];
  const float* W_x     = (const float*)d_in[5];
  const float* b       = (const float*)d_in[6];

  float* out_sec = (float*)d_out;                          // [T,B,D]
  float* h_sec   = out_sec + (size_t)T_STEPS * BD;         // [T+1,B,D]

  // workspace layout (~164 MiB used)
  char* ws = (char*)d_ws;
  unsigned* counter = (unsigned*)ws;                       // 256 B
  __bf16* hbuf = (__bf16*)(ws + 256);                      // 2*BD bf16 = 128 KiB
  __bf16* xb   = (__bf16*)(ws + 256 + 131072);             // T*B*D bf16
  __bf16* wab  = xb  + (size_t)TBROWS * DIM;               // D*D bf16
  __bf16* wxb  = wab + (size_t)DIM * DIM;                  // D*D bf16
  float* alpha_all = (float*)(wxb + (size_t)DIM * DIM);    // T*B*D f32
  float* wx_all    = alpha_all + (size_t)TBROWS * DIM;     // T*B*D f32

  // 1) convert X / W_alpha / W_x to bf16
  {
    int n4 = (TBROWS * DIM) / 4;                           // 4,194,304
    cvt4_kernel<<<n4 / 256, 256, 0, stream>>>(x, xb, n4);
    int w4 = (DIM * DIM) / 4;                              // 262,144
    cvt4_kernel<<<w4 / 256, 256, 0, stream>>>(W_alpha, wab, w4);
    cvt4_kernel<<<w4 / 256, 256, 0, stream>>>(W_x, wxb, w4);
  }

  // 2) init h[0], bf16 h buffer, barrier counter
  init_kernel<<<BD / 256, 256, 0, stream>>>(h0, h_sec, hbuf, counter);

  // 3) input GEMMs: 2 matrices * 1024 mtiles * 16 ngroups = 32768 waves
  input_gemm_kernel<<<4096, 256, 0, stream>>>(xb, wab, wxb, b_alpha,
                                              alpha_all, wx_all);

  // 4) persistent recurrent scan (64 KB static LDS per WG, 32 WGs x 128 thr)
  scan_kernel<<<NWG_SCAN, 128, 0, stream>>>(W_h, b, alpha_all, wx_all, h0,
                                            out_sec, h_sec, hbuf, counter);
}
